// FlashAttentionMPS_48421461295889
// MI455X (gfx1250) — matrix operand order repaired and checked by reading, not compiled, not run on hardware
//
#include <hip/hip_runtime.h>
#include <hip/hip_bf16.h>

// FlashAttention (block-diagonal causal), CDNA5/gfx1250, bf16 WMMA + fp32 accum,
// async global->LDS staging (ASYNCcnt) + double-buffered LDS pipeline.
// B=4, S=4096, E=1024, H=1024, BS=128, NB=32.
//
// Pipeline (all on `stream`):
//   1) qkv_gemm : qkv = x @ W_qkv            -> q,k,v bf16 in ws
//   2) attn_sm  : P = softmax(mask(q k^T/s)) -> P bf16 in ws
//   3) pv_gemm  : O = P @ v                  -> bf16, overwrites q region
//   4) out_proj : out = O @ W_out^T + b_out  -> fp32 d_out
// ws usage: 3*32MB (q,k,v) + 4MB (P) = 100 MB.

typedef __attribute__((ext_vector_type(16))) __bf16 v16bf;
typedef __attribute__((ext_vector_type(8)))  float  v8f;

static constexpr int RS = 40;  // LDS row stride in bf16 elems (80B: conflict-free frag loads)

union FragU { uint4 q[2]; v16bf v; };

// -------- CDNA5 async global->LDS copy (GLOBAL_LOAD_ASYNC_TO_LDS_B128) ------
// LDS byte address comes from vdst, global address from vaddr; both get the imm offset
// (IOFFSET is added on BOTH sides, so one base + offset:16 moves 32B/thread.)
#define ASYNC_COPY_B128(ldsoff, gaddr, imm)                                   \
  asm volatile("global_load_async_to_lds_b128 %0, %1, off offset:" #imm      \
               :: "v"(ldsoff), "v"(gaddr) : "memory")
#define WAIT_ASYNC(n) asm volatile("s_wait_asynccnt " #n ::: "memory")

__device__ __forceinline__ unsigned lds_off32(const void* p) {
  // generic pointers into the LDS aperture carry the LDS byte offset in the
  // low 32 bits (flat->LDS mapping truncates, ISA 10.2)
  return (unsigned)(unsigned long long)p;
}

__device__ __forceinline__ unsigned short f2bf(float f) {
  unsigned int x = __float_as_uint(f);
  x += 0x7FFFu + ((x >> 16) & 1u);       // round-to-nearest-even
  return (unsigned short)(x >> 16);
}

// ---- LDS staging helpers (tile = 128 rows x 32 K, 256 threads) --------------

// bf16 global rows -> LDS [row][k], pure DMA (2 async b128 per thread, ASYNCcnt += 2)
__device__ __forceinline__ void stage_bf16_async(unsigned short* lds,
                                                 const unsigned short* __restrict__ g,
                                                 int ldg, int t) {
  const int row = t >> 1, half = t & 1;
  const unsigned long long ga = (unsigned long long)(g + (size_t)row * ldg + half * 16);
  const unsigned lo = lds_off32(lds + row * RS + half * 16);
  ASYNC_COPY_B128(lo, ga, 0);
  ASYNC_COPY_B128(lo, ga, 16);
}

// fp32 global rows -> bf16 LDS [row][k] (conversion -> must go through VGPRs)
__device__ __forceinline__ void stage_f32(unsigned short* lds, const float* __restrict__ g,
                                          int ldg, int t) {
  const int row = t >> 1, half = t & 1;
  const float4* s = (const float4*)(g + (size_t)row * ldg + half * 16);
  float4 f0 = s[0], f1 = s[1], f2 = s[2], f3 = s[3];
  unsigned int p0 = f2bf(f0.x) | ((unsigned)f2bf(f0.y) << 16);
  unsigned int p1 = f2bf(f0.z) | ((unsigned)f2bf(f0.w) << 16);
  unsigned int p2 = f2bf(f1.x) | ((unsigned)f2bf(f1.y) << 16);
  unsigned int p3 = f2bf(f1.z) | ((unsigned)f2bf(f1.w) << 16);
  unsigned int p4 = f2bf(f2.x) | ((unsigned)f2bf(f2.y) << 16);
  unsigned int p5 = f2bf(f2.z) | ((unsigned)f2bf(f2.w) << 16);
  unsigned int p6 = f2bf(f3.x) | ((unsigned)f2bf(f3.y) << 16);
  unsigned int p7 = f2bf(f3.z) | ((unsigned)f2bf(f3.w) << 16);
  uint4* d = (uint4*)(lds + row * RS + half * 16);
  d[0] = make_uint4(p0, p1, p2, p3);
  d[1] = make_uint4(p4, p5, p6, p7);
}

// fp32 global [k][n] tile -> bf16 LDS [n][k] (transpose + convert)
__device__ __forceinline__ void stage_tr_f32(unsigned short* lds, const float* __restrict__ g,
                                             int ldg, int t) {
  const int k = t >> 3, seg = t & 7;
  const float4* s = (const float4*)(g + (size_t)k * ldg + seg * 16);
  float4 f0 = s[0], f1 = s[1], f2 = s[2], f3 = s[3];
  float f[16] = {f0.x, f0.y, f0.z, f0.w, f1.x, f1.y, f1.z, f1.w,
                 f2.x, f2.y, f2.z, f2.w, f3.x, f3.y, f3.z, f3.w};
  const int n0 = seg * 16;
#pragma unroll
  for (int j = 0; j < 16; ++j) lds[(n0 + j) * RS + k] = f2bf(f[j]);
}

// bf16 global [k][n] tile -> LDS [n][k] (transpose)
__device__ __forceinline__ void stage_tr_bf16(unsigned short* lds, const unsigned short* __restrict__ g,
                                              int ldg, int t) {
  const int k = t >> 3, seg = t & 7;
  const uint4* s = (const uint4*)(g + (size_t)k * ldg + seg * 16);
  union { uint4 q; unsigned short h[8]; } u0, u1;
  u0.q = s[0]; u1.q = s[1];
  const int n0 = seg * 16;
#pragma unroll
  for (int j = 0; j < 8; ++j) lds[(n0 + j) * RS + k] = u0.h[j];
#pragma unroll
  for (int j = 0; j < 8; ++j) lds[(n0 + 8 + j) * RS + k] = u1.h[j];
}

// ---- WMMA fragment loads (layouts per ISA 7.12.2, wave32) -------------------

// A 16x32 bf16: lane -> M = lane%16; elems e: K = (e>=8)*16 + (lane/16)*8 + e%8
__device__ __forceinline__ v16bf load_a(const unsigned short* lds, int m0, int lane) {
  const int m = m0 + (lane & 15);
  const int bk = (lane >> 4) << 3;
  FragU f;
  f.q[0] = *(const uint4*)(lds + m * RS + bk);
  f.q[1] = *(const uint4*)(lds + m * RS + bk + 16);
  return f.v;
}

// B 32x16 bf16 (LDS holds B^T as [n][k]): lane -> N = lane%16; K = (e>=8)*16 + (lane/16)*8 + e%8
__device__ __forceinline__ v16bf load_b(const unsigned short* lds, int n0, int lane) {
  const int n = n0 + (lane & 15);
  const int bk = (lane >> 4) << 3;
  FragU f;
  f.q[0] = *(const uint4*)(lds + n * RS + bk);
  f.q[1] = *(const uint4*)(lds + n * RS + bk + 16);
  return f.v;
}

__device__ __forceinline__ v8f wmma_bf16(v16bf a, v16bf b, v8f c) {
  // v_wmma_f32_16x16x32_bf16: (neg_a, A, neg_b, B, c_mod, C, reuse_a, reuse_b)
  return __builtin_amdgcn_wmma_f32_16x16x32_bf16(false, a, false, b, (short)0, c, false, false);
}

// C/D 16x16 f32: VGPR r, lanes0-15: (M=r, N=lane); lanes16-31: (M=r+8, N=lane-16)
__device__ __forceinline__ void store_tile_bf16(unsigned short* dst, int ld, int r0, int c0,
                                                const v8f& c, int lane) {
  const int col = c0 + (lane & 15);
  const int rb  = r0 + ((lane >> 4) << 3);
#pragma unroll
  for (int r = 0; r < 8; ++r) dst[(size_t)(rb + r) * ld + col] = f2bf(c[r]);
}

// ---- Kernel 1: qkv = x @ W_qkv (fp32 in, bf16 out) --------------------------
// Both operands need fp32->bf16 conversion -> manual staging, double-buffered.

__global__ __launch_bounds__(256) void qkv_gemm_kernel(
    const float* __restrict__ x, const float* __restrict__ Wqkv,
    unsigned short* __restrict__ q, unsigned short* __restrict__ k,
    unsigned short* __restrict__ v) {
  __shared__ unsigned short ldsA[2][128 * RS];
  __shared__ unsigned short ldsB[2][128 * RS];
  const int n0 = blockIdx.x * 128;        // 0..2944 over 3H
  const int m0 = blockIdx.y * 128;        // 0..16256 over B*S
  const int t = threadIdx.x, lane = t & 31, w = t >> 5;
  const int wm = (w >> 1) << 5;           // wave rows: 0/32/64/96
  const int wn = (w & 1) << 6;            // wave cols: 0/64
  const v8f vzero = {0.f, 0.f, 0.f, 0.f, 0.f, 0.f, 0.f, 0.f};
  v8f acc[2][4];
#pragma unroll
  for (int i = 0; i < 2; ++i)
#pragma unroll
    for (int j = 0; j < 4; ++j) acc[i][j] = vzero;

  stage_f32(ldsA[0], x + (size_t)m0 * 1024, 1024, t);
  stage_tr_f32(ldsB[0], Wqkv + n0, 3072, t);

  for (int i = 0; i < 32; ++i) {
    const int cur = i & 1, nxt = cur ^ 1;
    if (i < 31) {
      const int kk = (i + 1) * 32;
      stage_f32(ldsA[nxt], x + (size_t)m0 * 1024 + kk, 1024, t);
      stage_tr_f32(ldsB[nxt], Wqkv + (size_t)kk * 3072 + n0, 3072, t);
    }
    __syncthreads();
#pragma unroll
    for (int ii = 0; ii < 2; ++ii) {
      v16bf a = load_a(ldsA[cur], wm + ii * 16, lane);
#pragma unroll
      for (int j = 0; j < 4; ++j) {
        v16bf b = load_b(ldsB[cur], wn + j * 16, lane);
        acc[ii][j] = wmma_bf16(a, b, acc[ii][j]);
      }
    }
    __syncthreads();
  }

  unsigned short* dst = q;
  int c0 = n0;
  if (n0 >= 2048)      { dst = v; c0 = n0 - 2048; }
  else if (n0 >= 1024) { dst = k; c0 = n0 - 1024; }
#pragma unroll
  for (int i = 0; i < 2; ++i)
#pragma unroll
    for (int j = 0; j < 4; ++j)
      store_tile_bf16(dst, 1024, m0 + wm + i * 16, c0 + wn + j * 16, acc[i][j], lane);
}

// ---- Kernel 2: per-block scores + causal softmax -> P (bf16) ----------------
// Q and K tiles are raw bf16 copies -> fully async DMA staging, double-buffered.
// Wave w owns rows [w*16, w*16+16) x all 128 cols -> softmax is intra-wave.

__global__ __launch_bounds__(256) void attn_scores_kernel(
    const unsigned short* __restrict__ q, const unsigned short* __restrict__ kmat,
    unsigned short* __restrict__ P) {
  __shared__ unsigned short ldsQ[2][128 * RS];
  __shared__ unsigned short ldsK[2][128 * RS];
  const int blk = blockIdx.x;                 // 0..127 = b*NB + nb
  const size_t srow = (size_t)blk * 128;
  const unsigned short* qg = q    + srow * 1024;
  const unsigned short* kg = kmat + srow * 1024;
  const int t = threadIdx.x, lane = t & 31, w = t >> 5;

  const v8f vzero = {0.f, 0.f, 0.f, 0.f, 0.f, 0.f, 0.f, 0.f};
  v8f acc[8];
#pragma unroll
  for (int nt = 0; nt < 8; ++nt) acc[nt] = vzero;

  stage_bf16_async(ldsQ[0], qg, 1024, t);
  stage_bf16_async(ldsK[0], kg, 1024, t);     // LDS[n][d] = K[n][d] == (K^T) as [n][k]

  for (int i = 0; i < 32; ++i) {
    const int cur = i & 1, nxt = cur ^ 1;
    if (i < 31) {
      const int kk = (i + 1) * 32;
      stage_bf16_async(ldsQ[nxt], qg + kk, 1024, t);
      stage_bf16_async(ldsK[nxt], kg + kk, 1024, t);
      WAIT_ASYNC(4);                          // retire previous batch, keep new 4 in flight
    } else {
      WAIT_ASYNC(0);
    }
    __syncthreads();
    v16bf a = load_a(ldsQ[cur], w * 16, lane);
#pragma unroll
    for (int nt = 0; nt < 8; ++nt) {
      v16bf b = load_b(ldsK[cur], nt * 16, lane);
      acc[nt] = wmma_bf16(a, b, acc[nt]);
    }
    __syncthreads();
  }

  const float scale = 0.03125f;               // 1024^-0.5
  const int col0 = lane & 15;
  const int rsel = (lane >> 4) << 3;          // +8 for upper half-wave
  unsigned short* Pg = P + srow * 128;

#pragma unroll
  for (int r = 0; r < 8; ++r) {
    const int row = w * 16 + rsel + r;        // in-block row
    float s[8];
    float mx = -1e30f;
#pragma unroll
    for (int nt = 0; nt < 8; ++nt) {
      const int col = nt * 16 + col0;
      float val = acc[nt][r] * scale;
      if (col > row) val = -1e30f;            // causal mask within diagonal block
      s[nt] = val;
      mx = fmaxf(mx, val);
    }
#pragma unroll
    for (int off = 1; off < 16; off <<= 1) mx = fmaxf(mx, __shfl_xor(mx, off, 32));
    float sum = 0.f;
#pragma unroll
    for (int nt = 0; nt < 8; ++nt) {
      float p = __expf(s[nt] - mx);
      s[nt] = p;
      sum += p;
    }
#pragma unroll
    for (int off = 1; off < 16; off <<= 1) sum += __shfl_xor(sum, off, 32);
    const float inv = 1.0f / (sum + 1e-6f);
#pragma unroll
    for (int nt = 0; nt < 8; ++nt)
      Pg[(size_t)row * 128 + nt * 16 + col0] = f2bf(s[nt] * inv);
  }
}

// ---- Kernel 3: O = P @ v (bf16 in, bf16 out; O overwrites q region) ---------
// A (P) is a raw bf16 copy -> async DMA; B (V) needs transpose -> manual.

__global__ __launch_bounds__(256) void pv_gemm_kernel(
    const unsigned short* __restrict__ P, const unsigned short* __restrict__ v,
    unsigned short* __restrict__ O) {
  __shared__ unsigned short ldsA[2][128 * RS];
  __shared__ unsigned short ldsB[2][128 * RS];
  const int h0  = blockIdx.x * 128;           // 0..896 over H
  const int blk = blockIdx.y;                 // 0..127
  const size_t srow = (size_t)blk * 128;
  const unsigned short* Pg = P + srow * 128;
  const unsigned short* vg = v + srow * 1024 + h0;
  const int t = threadIdx.x, lane = t & 31, w = t >> 5;
  const int wm = (w >> 1) << 5;
  const int wn = (w & 1) << 6;
  const v8f vzero = {0.f, 0.f, 0.f, 0.f, 0.f, 0.f, 0.f, 0.f};
  v8f acc[2][4];
#pragma unroll
  for (int i = 0; i < 2; ++i)
#pragma unroll
    for (int j = 0; j < 4; ++j) acc[i][j] = vzero;

  stage_bf16_async(ldsA[0], Pg, 128, t);
  stage_tr_bf16(ldsB[0], vg, 1024, t);        // LDS[h][s] = V[s][h]

  for (int i = 0; i < 4; ++i) {
    const int cur = i & 1, nxt = cur ^ 1;
    if (i < 3) {
      const int kk = (i + 1) * 32;
      stage_bf16_async(ldsA[nxt], Pg + kk, 128, t);
      stage_tr_bf16(ldsB[nxt], vg + (size_t)kk * 1024, 1024, t);
      WAIT_ASYNC(2);
    } else {
      WAIT_ASYNC(0);
    }
    __syncthreads();
#pragma unroll
    for (int ii = 0; ii < 2; ++ii) {
      v16bf a = load_a(ldsA[cur], wm + ii * 16, lane);
#pragma unroll
      for (int j = 0; j < 4; ++j) {
        v16bf b = load_b(ldsB[cur], wn + j * 16, lane);
        acc[ii][j] = wmma_bf16(a, b, acc[ii][j]);
      }
    }
    __syncthreads();
  }

  unsigned short* Og = O + srow * 1024;
#pragma unroll
  for (int i = 0; i < 2; ++i)
#pragma unroll
    for (int j = 0; j < 4; ++j)
      store_tile_bf16(Og, 1024, wm + i * 16, h0 + wn + j * 16, acc[i][j], lane);
}

// ---- Kernel 4: out = O @ W_out^T + b_out (fp32 out) -------------------------
// A (O) async DMA; B = W_out rows [n][k] -> transpose-free but fp32 -> manual.

__global__ __launch_bounds__(256) void out_proj_kernel(
    const unsigned short* __restrict__ O, const float* __restrict__ Wout,
    const float* __restrict__ bout, float* __restrict__ out) {
  __shared__ unsigned short ldsA[2][128 * RS];
  __shared__ unsigned short ldsB[2][128 * RS];
  const int n0 = blockIdx.x * 128;            // 0..896 over H
  const int m0 = blockIdx.y * 128;            // 0..16256 over B*S
  const int t = threadIdx.x, lane = t & 31, w = t >> 5;
  const int wm = (w >> 1) << 5;
  const int wn = (w & 1) << 6;
  const v8f vzero = {0.f, 0.f, 0.f, 0.f, 0.f, 0.f, 0.f, 0.f};
  v8f acc[2][4];
#pragma unroll
  for (int i = 0; i < 2; ++i)
#pragma unroll
    for (int j = 0; j < 4; ++j) acc[i][j] = vzero;

  stage_bf16_async(ldsA[0], O + (size_t)m0 * 1024, 1024, t);
  stage_f32(ldsB[0], Wout + (size_t)n0 * 1024, 1024, t);

  for (int i = 0; i < 32; ++i) {
    const int cur = i & 1, nxt = cur ^ 1;
    if (i < 31) {
      const int kk = (i + 1) * 32;
      stage_bf16_async(ldsA[nxt], O + (size_t)m0 * 1024 + kk, 1024, t);
      stage_f32(ldsB[nxt], Wout + (size_t)n0 * 1024 + kk, 1024, t);
      WAIT_ASYNC(2);
    } else {
      WAIT_ASYNC(0);
    }
    __syncthreads();
#pragma unroll
    for (int ii = 0; ii < 2; ++ii) {
      v16bf a = load_a(ldsA[cur], wm + ii * 16, lane);
#pragma unroll
      for (int j = 0; j < 4; ++j) {
        v16bf b = load_b(ldsB[cur], wn + j * 16, lane);
        acc[ii][j] = wmma_bf16(a, b, acc[ii][j]);
      }
    }
    __syncthreads();
  }

#pragma unroll
  for (int i = 0; i < 2; ++i) {
#pragma unroll
    for (int j = 0; j < 4; ++j) {
      const int col = n0 + wn + j * 16 + (lane & 15);
      const int rb  = m0 + wm + i * 16 + ((lane >> 4) << 3);
      const float bias = bout[col];
#pragma unroll
      for (int r = 0; r < 8; ++r)
        out[(size_t)(rb + r) * 1024 + col] = acc[i][j][r] + bias;
    }
  }
}

// ---- host-side launch -------------------------------------------------------

extern "C" void kernel_launch(void* const* d_in, const int* in_sizes, int n_in,
                              void* d_out, int out_size, void* d_ws, size_t ws_size,
                              hipStream_t stream) {
  (void)in_sizes; (void)n_in; (void)out_size; (void)ws_size;
  const float* x    = (const float*)d_in[0];   // [4,4096,1024]
  const float* Wqkv = (const float*)d_in[1];   // [1024,3072]
  const float* Wout = (const float*)d_in[2];   // [1024,1024]
  const float* bout = (const float*)d_in[3];   // [1024]
  float* out = (float*)d_out;                  // [4,4096,1024] fp32

  // Workspace layout (bf16 = 2B): q | k | v | P   (O overwrites q after it's dead)
  const size_t NTOK = (size_t)4 * 4096;        // B*S = 16384
  unsigned short* q = (unsigned short*)d_ws;
  unsigned short* k = q + NTOK * 1024;
  unsigned short* v = k + NTOK * 1024;
  unsigned short* P = v + NTOK * 1024;         // [128 blocks][128][128]
  unsigned short* O = q;                       // reuse: q fully consumed by kernel 2

  qkv_gemm_kernel   <<<dim3(24, 128), 256, 0, stream>>>(x, Wqkv, q, k, v);
  attn_scores_kernel<<<dim3(128),     256, 0, stream>>>(q, k, P);
  pv_gemm_kernel    <<<dim3(8, 128),  256, 0, stream>>>(P, v, O);
  out_proj_kernel   <<<dim3(8, 128),  256, 0, stream>>>(O, Wout, bout, out);
}
